// MambaBlock_34857954574895
// MI455X (gfx1250) — compile-verified, matrix operand order checked by reading, not run on hardware
//
#include <hip/hip_runtime.h>
#include <hip/hip_bf16.h>

// ---------------- types for CDNA5 WMMA ----------------
typedef __attribute__((ext_vector_type(16))) __bf16 v16bf;
typedef __attribute__((ext_vector_type(8)))  float  v8f;

union FragBF16 {
    uint4 u4[2];   // 32 bytes
    v16bf v;
};

// ---------------- constants ----------------
#define BB      8
#define LL      2048
#define DM      768
#define DI      1536      // D_INNER
#define DS      16        // D_STATE
#define NTOK    (BB*LL)   // 16384

// ---------------- CDNA5 async global->LDS path (guarded) ----------------
#if defined(__has_builtin)
#if __has_builtin(__builtin_amdgcn_global_load_async_to_lds_b128) && \
    __has_builtin(__builtin_amdgcn_s_wait_asynccnt)
#define HAVE_ASYNC_LDS 1
#endif
#endif

#if defined(HAVE_ASYNC_LDS)
// builtin expects: (int4 addrspace(1)*, int4 addrspace(3)*, imm offset, imm cpol)
typedef int v4i_gcc __attribute__((vector_size(16)));
typedef __attribute__((address_space(1))) v4i_gcc as1_v4i;
typedef __attribute__((address_space(3))) v4i_gcc as3_v4i;
__device__ __forceinline__ void cp16_async(const unsigned short* g, unsigned short* l) {
    __builtin_amdgcn_global_load_async_to_lds_b128((as1_v4i*)g, (as3_v4i*)l, 0, 0);
}
#endif

__device__ __forceinline__ unsigned short f2bf(float f) {
    unsigned u = __float_as_uint(f);
    u += 0x7FFFu + ((u >> 16) & 1u);   // round-to-nearest-even
    return (unsigned short)(u >> 16);
}

// ---------------- f32 -> bf16 convert (with zero tail padding) ----------------
__global__ __launch_bounds__(256) void cvt_bf16_kernel(const float* __restrict__ src,
                                                       unsigned short* __restrict__ dst,
                                                       int n_src, int n_total) {
    int i = blockIdx.x * 256 + threadIdx.x;
    if (i < n_total) dst[i] = (i < n_src) ? f2bf(src[i]) : (unsigned short)0;
}

// ---------------- LayerNorm over 768, output bf16 ----------------
__global__ __launch_bounds__(256) void ln_kernel(const float* __restrict__ x,
                                                 const float* __restrict__ g,
                                                 const float* __restrict__ bta,
                                                 unsigned short* __restrict__ xnbf) {
    __shared__ float s1[256];
    __shared__ float s2[256];
    const int t   = blockIdx.x;            // token
    const int tid = threadIdx.x;
    const float* xr = x + (size_t)t * DM;
    float a0 = xr[tid], a1 = xr[tid + 256], a2 = xr[tid + 512];
    s1[tid] = a0 + a1 + a2;
    s2[tid] = a0*a0 + a1*a1 + a2*a2;
    __syncthreads();
    for (int off = 128; off > 0; off >>= 1) {
        if (tid < off) { s1[tid] += s1[tid+off]; s2[tid] += s2[tid+off]; }
        __syncthreads();
    }
    const float mu  = s1[0] * (1.0f / DM);
    const float var = s2[0] * (1.0f / DM) - mu * mu;
    const float inv = rsqrtf(var + 1e-5f);
    unsigned short* o = xnbf + (size_t)t * DM;
    o[tid      ] = f2bf((a0 - mu) * inv * g[tid      ] + bta[tid      ]);
    o[tid + 256] = f2bf((a1 - mu) * inv * g[tid + 256] + bta[tid + 256]);
    o[tid + 512] = f2bf((a2 - mu) * inv * g[tid + 512] + bta[tid + 512]);
}

// ---------------- generic bf16 WMMA GEMM:  C[M,N] = A[M,K] * B[N,K]^T (+resid) ----------------
// block: 256 threads (8 waves). block tile 128(M) x 64(N). wave tile 32x32.
// double-buffered LDS; global->LDS via CDNA5 async copies when available.
#define LDSROW 40   // padded row stride (ushorts) -> 80B, conflict-free, 16B aligned
__global__ __launch_bounds__(256) void gemm_bf16_wmma(const unsigned short* __restrict__ A,
                                                      const unsigned short* __restrict__ B,
                                                      float* __restrict__ C,
                                                      const float* __restrict__ resid,
                                                      int K, int lda, int ldb, int ldc) {
    __shared__ __align__(16) unsigned short As[2][128 * LDSROW];
    __shared__ __align__(16) unsigned short Bs[2][64  * LDSROW];

    const int tid  = threadIdx.x;
    const int m0   = blockIdx.y * 128;
    const int n0   = blockIdx.x * 64;
    const int wave = tid >> 5;
    const int lane = tid & 31;
    const int half = lane >> 4;       // 0/1
    const int l16  = lane & 15;
    const int wm   = wave & 3;        // 4 waves along M
    const int wn   = wave >> 2;       // 2 waves along N

    // per-thread staging chunks: A rows {r, r+64}, B row r; 8 bf16 (16B) each
    const int srow = tid >> 2;        // 0..63
    const int scol = (tid & 3) << 3;  // 0,8,16,24

    v8f acc[2][2];
    #pragma unroll
    for (int i = 0; i < 2; ++i)
        #pragma unroll
        for (int j = 0; j < 2; ++j)
            #pragma unroll
            for (int r = 0; r < 8; ++r) acc[i][j][r] = 0.0f;

    auto load_tile = [&](int s, int k0) {
        const unsigned short* ga0 = A + (size_t)(m0 + srow)      * lda + k0 + scol;
        const unsigned short* ga1 = A + (size_t)(m0 + srow + 64) * lda + k0 + scol;
        const unsigned short* gb  = B + (size_t)(n0 + srow)      * ldb + k0 + scol;
        unsigned short* la0 = &As[s][ srow       * LDSROW + scol];
        unsigned short* la1 = &As[s][(srow + 64) * LDSROW + scol];
        unsigned short* lb  = &Bs[s][ srow       * LDSROW + scol];
#if defined(HAVE_ASYNC_LDS)
        cp16_async(ga0, la0);
        cp16_async(ga1, la1);
        cp16_async(gb,  lb);
#else
        *(uint4*)la0 = *(const uint4*)ga0;
        *(uint4*)la1 = *(const uint4*)ga1;
        *(uint4*)lb  = *(const uint4*)gb;
#endif
    };

    load_tile(0, 0);
    int cur = 0;
    for (int k0 = 0; k0 < K; k0 += 32) {
#if defined(HAVE_ASYNC_LDS)
        __builtin_amdgcn_s_wait_asynccnt(0);
#endif
        __syncthreads();                 // stage `cur` visible; everyone done with `cur^1`
        if (k0 + 32 < K) load_tile(cur ^ 1, k0 + 32);

        FragBF16 fa[2], fb[2];
        #pragma unroll
        for (int i = 0; i < 2; ++i) {
            const uint4* p = (const uint4*)(&As[cur][(wm * 32 + i * 16 + l16) * LDSROW]);
            fa[i].u4[0] = p[half];
            fa[i].u4[1] = p[2 + half];
        }
        #pragma unroll
        for (int j = 0; j < 2; ++j) {
            const uint4* p = (const uint4*)(&Bs[cur][(wn * 32 + j * 16 + l16) * LDSROW]);
            fb[j].u4[0] = p[half];
            fb[j].u4[1] = p[2 + half];
        }
        #pragma unroll
        for (int i = 0; i < 2; ++i)
            #pragma unroll
            for (int j = 0; j < 2; ++j)
                acc[i][j] = __builtin_amdgcn_wmma_f32_16x16x32_bf16(
                    false, fa[i].v, false, fb[j].v, (short)0, acc[i][j], false, false);
        cur ^= 1;
    }

    // epilogue: D layout -> M = base + r + 8*half, N = base + l16
    #pragma unroll
    for (int i = 0; i < 2; ++i) {
        #pragma unroll
        for (int j = 0; j < 2; ++j) {
            const int mbase = m0 + wm * 32 + i * 16 + half * 8;
            const int n     = n0 + wn * 32 + j * 16 + l16;
            #pragma unroll
            for (int r = 0; r < 8; ++r) {
                size_t idx = (size_t)(mbase + r) * ldc + n;
                float v = acc[i][j][r];
                if (resid) v += resid[idx];
                C[idx] = v;
            }
        }
    }
}

// ---------------- depthwise causal conv (K=4) + bias + SiLU ----------------
__global__ __launch_bounds__(256) void conv_silu_kernel(const float* __restrict__ xz,   // [NTOK,3072], xb = cols 0..1535
                                                        const float* __restrict__ cw,   // [1536,1,4]
                                                        const float* __restrict__ cb,   // [1536]
                                                        float* __restrict__ xbc,        // [NTOK,1536]
                                                        unsigned short* __restrict__ xbcbf) {
    long long gid = (long long)blockIdx.x * 256 + threadIdx.x;
    if (gid >= (long long)NTOK * DI) return;
    int d = (int)(gid % DI);
    int t = (int)(gid / DI);
    int l = t % LL;
    int b = t / LL;
    float w0 = cw[d*4+0], w1 = cw[d*4+1], w2 = cw[d*4+2], w3 = cw[d*4+3];
    float acc = cb[d];
    const float* base = xz + ((size_t)b * LL) * (2*DI) + d;
    if (l - 3 >= 0) acc += w0 * base[(size_t)(l-3) * (2*DI)];
    if (l - 2 >= 0) acc += w1 * base[(size_t)(l-2) * (2*DI)];
    if (l - 1 >= 0) acc += w2 * base[(size_t)(l-1) * (2*DI)];
    acc += w3 * base[(size_t)l * (2*DI)];
    float v = acc / (1.0f + __expf(-acc));   // SiLU
    xbc[(size_t)t * DI + d]   = v;
    xbcbf[(size_t)t * DI + d] = f2bf(v);
}

// ---------------- selective scan + z-gate, output bf16 for out_proj ----------------
__global__ __launch_bounds__(256) void scan_kernel(const float* __restrict__ xp,    // [NTOK,64]: 0=dt_raw, 1..16=B, 17..32=C
                                                   const float* __restrict__ xbc,   // [NTOK,1536]
                                                   const float* __restrict__ xz,    // [NTOK,3072], z = cols 1536..
                                                   const float* __restrict__ dt_w,  // [1536]
                                                   const float* __restrict__ dt_b,  // [1536]
                                                   const float* __restrict__ A_log, // [1536,16]
                                                   const float* __restrict__ Dp,    // [1536]
                                                   unsigned short* __restrict__ ybf) {
    int g = blockIdx.x * 256 + threadIdx.x;       // 12288 = B * D_INNER
    int d = g % DI;
    int b = g / DI;
    float Av[DS];
    #pragma unroll
    for (int n = 0; n < DS; ++n) Av[n] = -__expf(A_log[(size_t)d * DS + n]);
    const float w  = dt_w[d];
    const float bb = dt_b[d];
    const float Dd = Dp[d];
    float h[DS];
    #pragma unroll
    for (int n = 0; n < DS; ++n) h[n] = 0.0f;

    for (int l = 0; l < LL; ++l) {
        const size_t t   = (size_t)b * LL + l;
        const float* row = xp + t * 64;
        float raw = row[0] * w + bb;
        float dt  = (raw > 20.0f) ? raw : log1pf(__expf(raw));
        float xt  = xbc[t * DI + d];
        float y   = 0.0f;
        #pragma unroll
        for (int n = 0; n < DS; ++n) {
            float dA = __expf(dt * Av[n]);
            h[n] = dA * h[n] + dt * row[1 + n] * xt;
            y += h[n] * row[17 + n];
        }
        y += Dd * xt;
        float zv   = xz[t * (2*DI) + DI + d];
        float gate = zv / (1.0f + __expf(-zv));
        ybf[t * DI + d] = f2bf(y * gate);
    }
}

// ---------------- host launcher ----------------
extern "C" void kernel_launch(void* const* d_in, const int* in_sizes, int n_in,
                              void* d_out, int out_size, void* d_ws, size_t ws_size,
                              hipStream_t stream) {
    const float* x        = (const float*)d_in[0];   // [8,2048,768]
    const float* ln_g     = (const float*)d_in[1];
    const float* ln_b     = (const float*)d_in[2];
    const float* in_w     = (const float*)d_in[3];   // [3072,768]
    const float* conv_w   = (const float*)d_in[4];   // [1536,1,4]
    const float* conv_b   = (const float*)d_in[5];
    const float* xp_w     = (const float*)d_in[6];   // [33,1536]
    const float* dt_w     = (const float*)d_in[7];   // [1536,1]
    const float* dt_b     = (const float*)d_in[8];
    const float* A_log    = (const float*)d_in[9];   // [1536,16]
    const float* Dvec     = (const float*)d_in[10];
    const float* out_w    = (const float*)d_in[11];  // [768,1536]
    float* out            = (float*)d_out;           // [8,2048,768]

    char* ws = (char*)d_ws;
    // workspace layout (all offsets 256B-aligned)
    size_t off = 0;
    unsigned short* wInBf  = (unsigned short*)(ws + off); off += (size_t)2*DI*DM*2;        // 3072x768 bf16
    unsigned short* wXpBf  = (unsigned short*)(ws + off); off += (size_t)64*DI*2;          // padded 64x1536 bf16
    unsigned short* wOutBf = (unsigned short*)(ws + off); off += (size_t)DM*DI*2;          // 768x1536 bf16
    unsigned short* xnBf   = (unsigned short*)(ws + off); off += (size_t)NTOK*DM*2;        // 16384x768 bf16
    float*          xz     = (float*)(ws + off);          off += (size_t)NTOK*2*DI*4;      // 16384x3072 f32
    float*          xbc    = (float*)(ws + off);          off += (size_t)NTOK*DI*4;        // 16384x1536 f32
    unsigned short* xbcBf  = (unsigned short*)(ws + off); off += (size_t)NTOK*DI*2;
    float*          xp     = (float*)(ws + off);          off += (size_t)NTOK*64*4;        // 16384x64 f32
    unsigned short* yBf    = (unsigned short*)(ws + off); off += (size_t)NTOK*DI*2;

    // 1) convert weights to bf16 (x_proj padded to 64 rows with zeros)
    {
        int n = 2*DI*DM;
        cvt_bf16_kernel<<<(n+255)/256, 256, 0, stream>>>(in_w, wInBf, n, n);
    }
    {
        int n_src = 33*DI, n_tot = 64*DI;
        cvt_bf16_kernel<<<(n_tot+255)/256, 256, 0, stream>>>(xp_w, wXpBf, n_src, n_tot);
    }
    {
        int n = DM*DI;
        cvt_bf16_kernel<<<(n+255)/256, 256, 0, stream>>>(out_w, wOutBf, n, n);
    }

    // 2) layernorm -> bf16
    ln_kernel<<<NTOK, 256, 0, stream>>>(x, ln_g, ln_b, xnBf);

    // 3) in_proj GEMM: [16384,768] x [3072,768]^T -> [16384,3072]
    gemm_bf16_wmma<<<dim3(2*DI/64, NTOK/128), 256, 0, stream>>>(
        xnBf, wInBf, xz, nullptr, DM, DM, DM, 2*DI);

    // 4) depthwise conv + SiLU
    {
        long long tot = (long long)NTOK * DI;
        conv_silu_kernel<<<(unsigned)((tot+255)/256), 256, 0, stream>>>(xz, conv_w, conv_b, xbc, xbcBf);
    }

    // 5) x_proj GEMM: [16384,1536] x [64,1536]^T -> [16384,64] (cols 0..32 valid)
    gemm_bf16_wmma<<<dim3(1, NTOK/128), 256, 0, stream>>>(
        xbcBf, wXpBf, xp, nullptr, DI, DI, DI, 64);

    // 6) selective scan + gate
    scan_kernel<<<(BB*DI)/256, 256, 0, stream>>>(xp, xbc, xz, dt_w, dt_b, A_log, Dvec, yBf);

    // 7) out_proj GEMM + residual: [16384,1536] x [768,1536]^T + x -> out
    gemm_bf16_wmma<<<dim3(DM/64, NTOK/128), 256, 0, stream>>>(
        yBf, wOutBf, out, x, DI, DI, DI, DM);
}
